// LSTMDisentangledAttention_15049565405471
// MI455X (gfx1250) — matrix operand order repaired and checked by reading, not compiled, not run on hardware
//
#include <hip/hip_runtime.h>
#include <hip/hip_bf16.h>
#include <math.h>

#define B_ 8
#define S_ 512
#define H_ 1024
#define NH_ 8
#define DH_ 128

typedef __attribute__((ext_vector_type(16))) __bf16 v16bf;
typedef __attribute__((ext_vector_type(8)))  float  v8f;
typedef __attribute__((ext_vector_type(4)))  unsigned u32x4;
typedef __attribute__((ext_vector_type(8)))  int      i32x8;
typedef __attribute__((ext_vector_type(4)))  int      i32x4;

#if defined(__has_builtin)
#if __has_builtin(__builtin_amdgcn_tensor_load_to_lds)
#define HAVE_TDM 1
#endif
#endif
#ifndef HAVE_TDM
#define HAVE_TDM 0
#endif

union FragU { v16bf v; unsigned u[8]; unsigned short s[16]; };
union AccU  { v8f v; float f[8]; };

__device__ __forceinline__ unsigned short f2bf(float f) {
  union { float f; unsigned u; } x; x.f = f;
  unsigned r = x.u + 0x7fffu + ((x.u >> 16) & 1u);
  return (unsigned short)(r >> 16);
}

__device__ __forceinline__ unsigned lds_off(const void* p) { return (unsigned)(size_t)p; }

// Async 16-byte global -> LDS copy (CDNA5 GLOBAL_LOAD_ASYNC_TO_LDS_B128, ASYNCcnt).
__device__ __forceinline__ void async_cp16(unsigned lds, const unsigned short* g) {
  asm volatile("global_load_async_to_lds_b128 %0, %1, off" :: "v"(lds), "v"(g) : "memory");
}
__device__ __forceinline__ void wait_async() {
  asm volatile("s_wait_asynccnt 0x0" ::: "memory");
}

// A fragment (16x32 bf16) from LDS tile [row][k]
__device__ __forceinline__ v16bf frag_a_lds(const unsigned short* t, int ld, int r0, int k0, int lane) {
  FragU f;
  int hi = (lane >> 4) & 1, m = lane & 15;
  const unsigned short* p = t + (r0 + m) * ld + k0;
  int ka = hi ? 16 : 0, kb = hi ? 24 : 8;
#pragma unroll
  for (int j = 0; j < 4; ++j) {
    f.u[j]     = *(const unsigned*)(p + ka + 2 * j);
    f.u[4 + j] = *(const unsigned*)(p + kb + 2 * j);
  }
  return f.v;
}
// B fragment (32x16 bf16) from LDS tile [n][k]
__device__ __forceinline__ v16bf frag_b_lds(const unsigned short* t, int ld, int n0, int k0, int lane) {
  FragU f;
  int hi = (lane >> 4) & 1, n = lane & 15;
  const unsigned short* p = t + (n0 + n) * ld + k0 + (hi ? 16 : 0);
#pragma unroll
  for (int j = 0; j < 8; ++j) f.u[j] = *(const unsigned*)(p + 2 * j);
  return f.v;
}
__device__ __forceinline__ v8f wmma_bf(v16bf a, v16bf b, v8f c) {
  return __builtin_amdgcn_wmma_f32_16x16x32_bf16(false, a, false, b, (short)0, c, false, false);
}

__global__ void cvt_bf16_kernel(const float* __restrict__ in, unsigned short* __restrict__ o, long n) {
  long i = (long)blockIdx.x * 256 + threadIdx.x;
  if (i < n) o[i] = f2bf(in[i]);
}

// ---------------------------------------------------------------------------
// Batched GEMM: C = epi( A[M,K]_bf16 * B[N,K]_bf16^T ). f32 acc.
// Staging = raw 16B async copies into LDS. Output f32 (C) and/or bf16 (Cb).
// M,N multiples of 128; K multiple of 32.
// ---------------------------------------------------------------------------
__global__ __launch_bounds__(256) void gemm_kernel(
    const unsigned short* __restrict__ A, const unsigned short* __restrict__ Bm,
    float* __restrict__ C, unsigned short* __restrict__ Cb,
    const float* __restrict__ bias, const float* __restrict__ resid,
    int K, int lda, int ldb, int ldc, float alpha, int act,
    int nb2, long sAb1, long sAb2, long sBb1, long sBb2, long sCb1, long sCb2)
{
  __shared__ unsigned short As[128 * 32];
  __shared__ unsigned short Bs[128 * 32];
  int tid = threadIdx.x, lane = tid & 31, wave = tid >> 5;
  int n0 = blockIdx.x * 128, m0 = blockIdx.y * 128;
  int bz = blockIdx.z, b1 = bz / nb2, b2 = bz % nb2;
  const unsigned short* Ap = A + (long)b1 * sAb1 + (long)b2 * sAb2;
  const unsigned short* Bp = Bm + (long)b1 * sBb1 + (long)b2 * sBb2;
  float* Cp = C ? C + (long)b1 * sCb1 + (long)b2 * sCb2 : (float*)0;
  unsigned short* Cbp = Cb ? Cb + (long)b1 * sCb1 + (long)b2 * sCb2 : (unsigned short*)0;
  const float* Rp = resid ? (resid + (long)b1 * sCb1 + (long)b2 * sCb2) : (const float*)0;

  int wm = (wave & 3) * 32, wn = (wave >> 2) * 64;
  const v8f vz = {};
  v8f acc[2][4];
#pragma unroll
  for (int i = 0; i < 2; ++i)
#pragma unroll
    for (int j = 0; j < 4; ++j) acc[i][j] = vz;

  for (int kk = 0; kk < K; kk += 32) {
    // 128 rows x 64B per tile = 512 chunks; 2 per thread per tile
    for (int ch = tid; ch < 512; ch += 256) {
      int r = ch >> 2, c = (ch & 3) * 8;
      async_cp16(lds_off(&As[r * 32 + c]), Ap + (long)(m0 + r) * lda + kk + c);
      async_cp16(lds_off(&Bs[r * 32 + c]), Bp + (long)(n0 + r) * ldb + kk + c);
    }
    if (kk + 32 < K)
      __builtin_prefetch(Ap + (long)(m0 + (tid >> 1)) * lda + kk + 32, 0, 1);
    wait_async();
    __syncthreads();
    v16bf af[2], bfg[4];
#pragma unroll
    for (int i = 0; i < 2; ++i) af[i] = frag_a_lds(As, 32, wm + 16 * i, 0, lane);
#pragma unroll
    for (int j = 0; j < 4; ++j) bfg[j] = frag_b_lds(Bs, 32, wn + 16 * j, 0, lane);
#pragma unroll
    for (int i = 0; i < 2; ++i)
#pragma unroll
      for (int j = 0; j < 4; ++j)
        acc[i][j] = wmma_bf(af[i], bfg[j], acc[i][j]);
    __syncthreads();
  }

  int hi = (lane >> 4) & 1, nl = lane & 15;
#pragma unroll
  for (int i = 0; i < 2; ++i)
#pragma unroll
    for (int j = 0; j < 4; ++j) {
      AccU a; a.v = acc[i][j];
#pragma unroll
      for (int e = 0; e < 8; ++e) {
        int row = m0 + wm + 16 * i + e + hi * 8;
        int col = n0 + wn + 16 * j + nl;
        float v = a.f[e];
        if (bias) v += bias[col];
        v *= alpha;
        if (Rp) v += Rp[(long)row * ldc + col];
        if (act == 1) v = 0.5f * v * (1.0f + erff(v * 0.70710678118654752f));
        if (Cp)  Cp[(long)row * ldc + col] = v;
        if (Cbp) Cbp[(long)row * ldc + col] = f2bf(v);
      }
    }
}

// Stage a [128 rows x 128 bf16] slice of pos tensor into LDS via the Tensor Data
// Mover (2D descriptor, rows past rows_avail read as zero). Fallback: manual copy.
__device__ __forceinline__ void stage_pos_tile(unsigned short* Ps, const unsigned short* g,
                                               int rows_avail, int tid) {
#if HAVE_TDM
  if (tid < 32) {  // wave 0 issues one TDM descriptor for the whole tile
    unsigned long long ga = (unsigned long long)(size_t)g;
    u32x4 g0; i32x8 g1; i32x4 gz = {0, 0, 0, 0}; i32x8 gz8 = {0, 0, 0, 0, 0, 0, 0, 0};
    g0[0] = 1u;                                                 // count=1
    g0[1] = lds_off(Ps);                                        // lds_addr
    g0[2] = (unsigned)ga;                                       // global_addr[31:0]
    g0[3] = (unsigned)((ga >> 32) & 0x01ffffffull) | (2u << 30); // ga[56:32] | type=2
    g1[0] = 1 << 16;                                            // data_size = 2 bytes
    g1[1] = (int)(128u << 16);                                  // tensor_dim0 lo16 (=128)
    g1[2] = (int)(((unsigned)rows_avail & 0xffffu) << 16);      // tdim0 hi | tensor_dim1 lo
    g1[3] = (int)(128u << 16);                                  // tdim1 hi | tile_dim0=128
    g1[4] = 128;                                                // tile_dim1=128, tile_dim2=0
    g1[5] = 1024;                                               // tensor_dim0_stride (elems)
    g1[6] = 0;
    g1[7] = 0;
    __builtin_amdgcn_tensor_load_to_lds(g0, g1, gz, gz, gz8, 0);
    __builtin_amdgcn_s_wait_tensorcnt(0);
  }
#else
  for (int i = tid; i < 128 * 128; i += 256) {
    int r = i >> 7, c = i & 127;
    int rr = r < rows_avail ? r : (rows_avail - 1);
    Ps[i] = g[(long)rr * 1024 + c];
  }
#endif
}

// ---------------------------------------------------------------------------
// Fused disentangled-attention scores: S = q.k^T + c2p + p2c per 64x64 tile.
// d = q-k+512 in [1,1023]; per tile d spans 127 values -> one 128-row pos slice.
// ---------------------------------------------------------------------------
__global__ __launch_bounds__(256) void scores_kernel(
    const unsigned short* __restrict__ qb, const unsigned short* __restrict__ kb,
    const unsigned short* __restrict__ poskb, const unsigned short* __restrict__ posqb,
    float* __restrict__ scores)
{
  __shared__ unsigned short Qs[64 * 128];
  __shared__ unsigned short Ks[64 * 128];
  __shared__ unsigned short Ps[128 * 128];
  __shared__ float Ss[64 * 64];
  int tid = threadIdx.x, lane = tid & 31, wave = tid >> 5;
  int k0 = blockIdx.x * 64, q0 = blockIdx.y * 64, bh = blockIdx.z;
  int b = bh >> 3, h = bh & 7;
  // async-stage Q and K tiles (raw bf16, 16B chunks)
  for (int ch = tid; ch < 1024; ch += 256) {
    int r = ch >> 4, c = (ch & 15) * 8;
    async_cp16(lds_off(&Qs[r * 128 + c]), qb + ((long)(b * S_ + q0 + r)) * H_ + h * DH_ + c);
    async_cp16(lds_off(&Ks[r * 128 + c]), kb + ((long)(b * S_ + k0 + r)) * H_ + h * DH_ + c);
  }
  wait_async();
  __syncthreads();

  int rbase = q0 - k0 + 449;  // d-window [rbase, rbase+126]
  // kick TDM load of pos_key slice; overlaps with phase 0 compute
  stage_pos_tile(Ps, poskb + ((long)rbase * H_ + h * DH_), 1024 - rbase, tid);

  int hi = (lane >> 4) & 1, nl = lane & 15;
  const v8f vz = {};
  // phase 0: q.k^T
  for (int t = wave * 2; t < wave * 2 + 2; ++t) {
    int mi = (t >> 2) * 16, ni = (t & 3) * 16;
    v8f acc = vz;
#pragma unroll
    for (int ks = 0; ks < 128; ks += 32)
      acc = wmma_bf(frag_a_lds(Qs, 128, mi, ks, lane), frag_b_lds(Ks, 128, ni, ks, lane), acc);
    AccU a; a.v = acc;
#pragma unroll
    for (int e = 0; e < 8; ++e) Ss[(mi + e + hi * 8) * 64 + ni + nl] = a.f[e];
  }
  __syncthreads();
  // phase 1: c2p = Q . PK^T, diagonal scatter-add (disjoint addresses)
  for (int t = wave * 4; t < wave * 4 + 4; ++t) {
    int mi = (t >> 3) * 16, di = (t & 7) * 16;
    v8f acc = vz;
#pragma unroll
    for (int ks = 0; ks < 128; ks += 32)
      acc = wmma_bf(frag_a_lds(Qs, 128, mi, ks, lane), frag_b_lds(Ps, 128, di, ks, lane), acc);
    AccU a; a.v = acc;
#pragma unroll
    for (int e = 0; e < 8; ++e) {
      int m = mi + e + hi * 8, dl = di + nl;
      int kkx = m + 63 - dl;
      if (kkx >= 0 && kkx < 64) Ss[m * 64 + kkx] += a.f[e];
    }
  }
  __syncthreads();
  stage_pos_tile(Ps, posqb + ((long)rbase * H_ + h * DH_), 1024 - rbase, tid);
  __syncthreads();
  // phase 2: p2c = K . PQ^T, transposed scatter-add
  for (int t = wave * 4; t < wave * 4 + 4; ++t) {
    int mi = (t >> 3) * 16, di = (t & 7) * 16;
    v8f acc = vz;
#pragma unroll
    for (int ks = 0; ks < 128; ks += 32)
      acc = wmma_bf(frag_a_lds(Ks, 128, mi, ks, lane), frag_b_lds(Ps, 128, di, ks, lane), acc);
    AccU a; a.v = acc;
#pragma unroll
    for (int e = 0; e < 8; ++e) {
      int km = mi + e + hi * 8, dl = di + nl;
      int qq = km + dl - 63;
      if (qq >= 0 && qq < 64) Ss[qq * 64 + km] += a.f[e];
    }
  }
  __syncthreads();
  float* out = scores + ((long)bh * S_ + q0) * S_ + k0;
  for (int i = tid; i < 64 * 64; i += 256) {
    int r = i >> 6, c = i & 63;
    out[(long)r * S_ + c] = Ss[i];
  }
}

// softmax row (f32 in) -> bf16 probs out; one wave per 512-wide row
__global__ void softmax_kernel(const float* __restrict__ scores, unsigned short* __restrict__ probs) {
  const float* p = scores + (long)blockIdx.x * S_;
  unsigned short* o = probs + (long)blockIdx.x * S_;
  int lane = threadIdx.x;
  float v[16], mx = -1e30f;
#pragma unroll
  for (int i = 0; i < 16; ++i) { v[i] = p[lane + 32 * i]; mx = fmaxf(mx, v[i]); }
  for (int off = 16; off; off >>= 1) mx = fmaxf(mx, __shfl_xor(mx, off));
  float s = 0.f;
#pragma unroll
  for (int i = 0; i < 16; ++i) { v[i] = __expf(v[i] - mx); s += v[i]; }
  for (int off = 16; off; off >>= 1) s += __shfl_xor(s, off);
  float inv = 1.0f / s;
#pragma unroll
  for (int i = 0; i < 16; ++i) o[lane + 32 * i] = f2bf(v[i] * inv);
}

__device__ __forceinline__ void ln_body(const float* loc0, float mean, float inv,
                                        const float* g, const float* bt,
                                        float* orow, unsigned short* obrow, int tid) {
#pragma unroll
  for (int i = 0; i < 4; ++i) {
    int c = tid + 256 * i;
    float v = (loc0[i] - mean) * inv * g[c] + bt[c];
    if (orow) orow[c] = v;
    if (obrow) obrow[c] = f2bf(v);
  }
}

__global__ __launch_bounds__(256) void embed_ln_kernel(
    const float* __restrict__ x, const float* __restrict__ pos,
    const float* __restrict__ g, const float* __restrict__ bt,
    float* __restrict__ out, unsigned short* __restrict__ outb)
{
  __shared__ float red[256];
  int row = blockIdx.x, s = row & (S_ - 1), tid = threadIdx.x;
  const float* xr = x + (long)row * H_;
  const float* pr = pos + (long)s * H_;
  float loc[4], sum = 0.f;
#pragma unroll
  for (int i = 0; i < 4; ++i) { loc[i] = xr[tid + 256 * i] + pr[tid + 256 * i]; sum += loc[i]; }
  red[tid] = sum; __syncthreads();
  for (int o = 128; o; o >>= 1) { if (tid < o) red[tid] += red[tid + o]; __syncthreads(); }
  float mean = red[0] * (1.0f / H_); __syncthreads();
  float sq = 0.f;
#pragma unroll
  for (int i = 0; i < 4; ++i) { float d = loc[i] - mean; sq += d * d; }
  red[tid] = sq; __syncthreads();
  for (int o = 128; o; o >>= 1) { if (tid < o) red[tid] += red[tid + o]; __syncthreads(); }
  float inv = rsqrtf(red[0] * (1.0f / H_) + 1e-7f);
  ln_body(loc, mean, inv, g, bt, out + (long)row * H_, outb + (long)row * H_, tid);
}

__global__ __launch_bounds__(256) void ln_kernel(
    const float* __restrict__ X, const float* __restrict__ g, const float* __restrict__ bt,
    float* __restrict__ out, unsigned short* __restrict__ outb)
{
  __shared__ float red[256];
  int row = blockIdx.x, tid = threadIdx.x;
  const float* xr = X + (long)row * H_;
  float loc[4], sum = 0.f;
#pragma unroll
  for (int i = 0; i < 4; ++i) { loc[i] = xr[tid + 256 * i]; sum += loc[i]; }
  red[tid] = sum; __syncthreads();
  for (int o = 128; o; o >>= 1) { if (tid < o) red[tid] += red[tid + o]; __syncthreads(); }
  float mean = red[0] * (1.0f / H_); __syncthreads();
  float sq = 0.f;
#pragma unroll
  for (int i = 0; i < 4; ++i) { float d = loc[i] - mean; sq += d * d; }
  red[tid] = sq; __syncthreads();
  for (int o = 128; o; o >>= 1) { if (tid < o) red[tid] += red[tid + o]; __syncthreads(); }
  float inv = rsqrtf(red[0] * (1.0f / H_) + 1e-7f);
  ln_body(loc, mean, inv, g, bt, out + (long)row * H_, outb + (long)row * H_, tid);
}

// q,v bias/scale fixup; emit bf16 q,k and pre-transposed v^T per (b,h)
__global__ void qkv_fixup_kernel(const float* __restrict__ qkv,
                                 const float* __restrict__ qbias, const float* __restrict__ vbias,
                                 unsigned short* __restrict__ qb, unsigned short* __restrict__ kb,
                                 unsigned short* __restrict__ vbt) {
  long i = (long)blockIdx.x * blockDim.x + threadIdx.x;   // over 4096*1024
  long row = i >> 10; int c = (int)(i & 1023);
  const float invs = 0.051031036307982884f;               // 1/sqrt(3*128)
  const float* base = qkv + row * (3 * H_);
  qb[row * H_ + c] = f2bf((base[c] + qbias[c]) * invs);
  kb[row * H_ + c] = f2bf(base[H_ + c]);
  float vv = base[2 * H_ + c] + vbias[c];
  long bb = row >> 9; int s = (int)(row & 511);
  int hh = c >> 7, dh = c & 127;
  vbt[(((bb * NH_ + hh) * DH_) + dh) * (long)S_ + s] = f2bf(vv);  // [b][h][dh][s]
}

__global__ void bias2_kernel(const float* __restrict__ a, const float* __restrict__ b,
                             float* __restrict__ o, int n) {
  int i = blockIdx.x * blockDim.x + threadIdx.x;
  if (i < n) o[i] = a[i] + b[i];
}

// ---------------------------------------------------------------------------
// Bidirectional LSTM: 1 workgroup/dir; h.Whh^T recurrence on WMMA (M 8->16 pad).
// ---------------------------------------------------------------------------
__global__ __launch_bounds__(256) void lstm_kernel(
    const float* __restrict__ xwF, const float* __restrict__ xwR,
    const unsigned short* __restrict__ WhhFb, const unsigned short* __restrict__ WhhRb,
    float* __restrict__ out)
{
  __shared__ float cS[8 * 512];
  __shared__ float gS[8 * 2048];
  __shared__ unsigned short hB[16 * 512];
  int dir = blockIdx.x;
  const float* xw = dir ? xwR : xwF;
  const unsigned short* Whh = dir ? WhhRb : WhhFb;
  int tid = threadIdx.x, lane = tid & 31, wave = tid >> 5;
  for (int i = tid; i < 8 * 512; i += 256) cS[i] = 0.f;
  for (int i = tid; i < 16 * 512; i += 256) hB[i] = 0;
  __syncthreads();
  int hi = (lane >> 4) & 1, nl = lane & 15;
  const v8f vz = {};
  for (int t = 0; t < S_; ++t) {
    int s = dir ? (S_ - 1 - t) : t;
    v16bf af[16];
#pragma unroll
    for (int ks = 0; ks < 16; ++ks) af[ks] = frag_a_lds(hB, 512, 0, ks * 32, lane);
    int nbase = wave * 256;
    for (int nt = 0; nt < 16; ++nt) {
      int ncol = nbase + nt * 16;
      v8f acc = vz;
      for (int ks = 0; ks < 16; ++ks) {
        FragU fb;  // raw bf16 B frag from global Whh [2048x512]
        const unsigned short* wp = Whh + (long)(ncol + nl) * 512 + ks * 32 + (hi ? 16 : 0);
#pragma unroll
        for (int j = 0; j < 8; ++j) fb.u[j] = *(const unsigned*)(wp + 2 * j);
        acc = wmma_bf(af[ks], fb.v, acc);
      }
      AccU a; a.v = acc;
#pragma unroll
      for (int e = 0; e < 8; ++e) {
        int m = e + hi * 8;
        if (m < 8) {
          int col = ncol + nl;
          gS[m * 2048 + col] = a.f[e] + xw[((long)m * S_ + s) * 2048 + col];
        }
      }
    }
    __syncthreads();
    for (int i = tid; i < 8 * 512; i += 256) {
      int bb = i >> 9, j = i & 511;
      const float* gr = gS + bb * 2048;
      float ig = 1.f / (1.f + __expf(-gr[j]));
      float fg = 1.f / (1.f + __expf(-gr[512 + j]));
      float gg = tanhf(gr[1024 + j]);
      float og = 1.f / (1.f + __expf(-gr[1536 + j]));
      float c  = fg * cS[i] + ig * gg;
      float hv = og * tanhf(c);
      cS[i] = c;
      hB[bb * 512 + j] = f2bf(hv);
      out[((long)bb * S_ + s) * H_ + dir * 512 + j] = hv;
    }
    __syncthreads();
  }
}

// ---------------------------------------------------------------------------
extern "C" void kernel_launch(void* const* d_in, const int* in_sizes, int n_in,
                              void* d_out, int out_size, void* d_ws, size_t ws_size,
                              hipStream_t stream) {
  (void)in_sizes; (void)n_in; (void)out_size; (void)ws_size;
  const float* x        = (const float*)d_in[0];
  const float* pos_emb  = (const float*)d_in[1];
  const float* eg       = (const float*)d_in[2];
  const float* eb       = (const float*)d_in[3];
  const float* in_proj  = (const float*)d_in[4];
  const float* q_bias   = (const float*)d_in[5];
  const float* v_bias   = (const float*)d_in[6];
  const float* rel_emb  = (const float*)d_in[7];
  const float* pos_proj = (const float*)d_in[8];
  const float* posq_w   = (const float*)d_in[9];
  const float* posq_b   = (const float*)d_in[10];
  const float* ao_w     = (const float*)d_in[11];
  const float* ao_b     = (const float*)d_in[12];
  const float* alg      = (const float*)d_in[13];
  const float* alb      = (const float*)d_in[14];
  const float* in_w     = (const float*)d_in[15];
  const float* in_b     = (const float*)d_in[16];
  const float* fo_w     = (const float*)d_in[17];
  const float* fo_b     = (const float*)d_in[18];
  const float* flg      = (const float*)d_in[19];
  const float* flb      = (const float*)d_in[20];
  const float* WihF     = (const float*)d_in[21];
  const float* WhhF     = (const float*)d_in[22];
  const float* bihF     = (const float*)d_in[23];
  const float* bhhF     = (const float*)d_in[24];
  const float* WihR     = (const float*)d_in[25];
  const float* WhhR     = (const float*)d_in[26];
  const float* bihR     = (const float*)d_in[27];
  const float* bhhR     = (const float*)d_in[28];
  float* out = (float*)d_out;

  // ---- workspace layout ----
  float* wsf = (float*)d_ws;
  size_t o = 0;
  float* h1     = wsf + o; o += (size_t)4096 * 1024;
  float* qkv    = wsf + o; o += (size_t)4096 * 3072;
  float* scores = wsf + o; o += (size_t)64 * 512 * 512;    // reused as xwf|xwr
  float* tbuf   = wsf + o; o += (size_t)4096 * 1024;
  float* h2     = wsf + o; o += (size_t)4096 * 1024;
  float* h3     = wsf + o; o += (size_t)4096 * 1024;
  float* biasf  = wsf + o; o += 2048;
  float* biasr  = wsf + o; o += 2048;
  float* xwf = scores;
  float* xwr = scores + (size_t)4096 * 2048;

  unsigned short* wsb = (unsigned short*)(wsf + o);
  size_t p = 0;
  unsigned short* in_projb = wsb + p; p += (size_t)3072 * 1024;
  unsigned short* rel_b    = wsb + p; p += (size_t)1024 * 1024;
  unsigned short* pprojb   = wsb + p; p += (size_t)1024 * 1024;
  unsigned short* pqwb     = wsb + p; p += (size_t)1024 * 1024;
  unsigned short* aowb     = wsb + p; p += (size_t)1024 * 1024;
  unsigned short* inwb     = wsb + p; p += (size_t)3072 * 1024;
  unsigned short* fowb     = wsb + p; p += (size_t)3072 * 1024;
  unsigned short* wihfb    = wsb + p; p += (size_t)2048 * 1024;
  unsigned short* wihrb    = wsb + p; p += (size_t)2048 * 1024;
  unsigned short* whhfb    = wsb + p; p += (size_t)2048 * 512;
  unsigned short* whhrb    = wsb + p; p += (size_t)2048 * 512;
  unsigned short* h1b      = wsb + p; p += (size_t)4096 * 1024;
  unsigned short* qb       = wsb + p; p += (size_t)4096 * 1024;
  unsigned short* kb       = wsb + p; p += (size_t)4096 * 1024;
  unsigned short* vbt      = wsb + p; p += (size_t)4096 * 1024;
  unsigned short* poskb    = wsb + p; p += (size_t)1024 * 1024;
  unsigned short* posqb    = wsb + p; p += (size_t)1024 * 1024;
  unsigned short* probsb   = wsb + p; p += (size_t)64 * 512 * 512;
  unsigned short* ctxb     = wsb + p; p += (size_t)4096 * 1024;
  unsigned short* h2b      = wsb + p; p += (size_t)4096 * 1024;
  unsigned short* interb   = wsb + p; p += (size_t)4096 * 3072;
  unsigned short* h3b      = wsb + p; p += (size_t)4096 * 1024;

  dim3 blk(256);
  auto cvt = [&](const float* src, unsigned short* dst, long n) {
    cvt_bf16_kernel<<<dim3((unsigned)((n + 255) / 256)), blk, 0, stream>>>(src, dst, n);
  };
  // one-time bf16 weight conversion (stays hot in 192MB L2)
  cvt(in_proj, in_projb, (long)3072 * 1024);
  cvt(rel_emb, rel_b,    (long)1024 * 1024);
  cvt(pos_proj, pprojb,  (long)1024 * 1024);
  cvt(posq_w, pqwb,      (long)1024 * 1024);
  cvt(ao_w, aowb,        (long)1024 * 1024);
  cvt(in_w, inwb,        (long)3072 * 1024);
  cvt(fo_w, fowb,        (long)3072 * 1024);
  cvt(WihF, wihfb,       (long)2048 * 1024);
  cvt(WihR, wihrb,       (long)2048 * 1024);
  cvt(WhhF, whhfb,       (long)2048 * 512);
  cvt(WhhR, whhrb,       (long)2048 * 512);

  // 1) h1 = LN(x + pos_emb)
  embed_ln_kernel<<<4096, blk, 0, stream>>>(x, pos_emb, eg, eb, h1, h1b);
  // 2) qkv = h1 @ in_proj^T
  gemm_kernel<<<dim3(24, 32, 1), blk, 0, stream>>>(h1b, in_projb, qkv, nullptr, nullptr, nullptr,
      1024, 1024, 1024, 3072, 1.0f, 0, 1, 0, 0, 0, 0, 0, 0);
  // 3) fixup -> bf16 q (scaled), k, v^T
  qkv_fixup_kernel<<<16384, blk, 0, stream>>>(qkv, q_bias, v_bias, qb, kb, vbt);
  // 4) pos_key / pos_q (bf16 out)
  gemm_kernel<<<dim3(8, 8, 1), blk, 0, stream>>>(rel_b, pprojb, nullptr, poskb, nullptr, nullptr,
      1024, 1024, 1024, 1024, 1.0f, 0, 1, 0, 0, 0, 0, 0, 0);
  gemm_kernel<<<dim3(8, 8, 1), blk, 0, stream>>>(rel_b, pqwb, nullptr, posqb, posq_b, nullptr,
      1024, 1024, 1024, 1024, 0.051031036307982884f, 0, 1, 0, 0, 0, 0, 0, 0);
  // 5) fused scores + softmax -> bf16 probs
  scores_kernel<<<dim3(8, 8, 64), blk, 0, stream>>>(qb, kb, poskb, posqb, scores);
  softmax_kernel<<<64 * 512, dim3(32), 0, stream>>>(scores, probsb);
  // 6) ctx = probs @ v  (batched b,h; bf16 out)
  gemm_kernel<<<dim3(1, 4, 64), blk, 0, stream>>>(probsb, vbt, nullptr, ctxb, nullptr, nullptr,
      512, 512, 512, 1024, 1.0f, 0, 8,
      (long)8 * 512 * 512, (long)512 * 512,
      (long)8 * 128 * 512, (long)128 * 512,
      (long)512 * 1024, 128);
  // 7) attn out + residual, LN
  gemm_kernel<<<dim3(8, 32, 1), blk, 0, stream>>>(ctxb, aowb, tbuf, nullptr, ao_b, h1,
      1024, 1024, 1024, 1024, 1.0f, 0, 1, 0, 0, 0, 0, 0, 0);
  ln_kernel<<<4096, blk, 0, stream>>>(tbuf, alg, alb, h2, h2b);
  // 8) FFN
  gemm_kernel<<<dim3(24, 32, 1), blk, 0, stream>>>(h2b, inwb, nullptr, interb, in_b, nullptr,
      1024, 1024, 1024, 3072, 1.0f, 1, 1, 0, 0, 0, 0, 0, 0);
  gemm_kernel<<<dim3(8, 32, 1), blk, 0, stream>>>(interb, fowb, tbuf, nullptr, fo_b, h2,
      3072, 3072, 3072, 1024, 1.0f, 0, 1, 0, 0, 0, 0, 0, 0);
  ln_kernel<<<4096, blk, 0, stream>>>(tbuf, flg, flb, h3, h3b);
  // 9) LSTM input projections (+combined biases)
  bias2_kernel<<<8, blk, 0, stream>>>(bihF, bhhF, biasf, 2048);
  bias2_kernel<<<8, blk, 0, stream>>>(bihR, bhhR, biasr, 2048);
  gemm_kernel<<<dim3(16, 32, 1), blk, 0, stream>>>(h3b, wihfb, xwf, nullptr, biasf, nullptr,
      1024, 1024, 1024, 2048, 1.0f, 0, 1, 0, 0, 0, 0, 0, 0);
  gemm_kernel<<<dim3(16, 32, 1), blk, 0, stream>>>(h3b, wihrb, xwr, nullptr, biasr, nullptr,
      1024, 1024, 1024, 2048, 1.0f, 0, 1, 0, 0, 0, 0, 0, 0);
  // 10) bidirectional LSTM -> d_out
  lstm_kernel<<<dim3(2), blk, 0, stream>>>(xwf, xwr, whhfb, whhrb, out);
}
